// DFConv2d_56616258896456
// MI455X (gfx1250) — matrix operand order repaired and checked by reading, not compiled, not run on hardware
//
#include <hip/hip_runtime.h>

// ---------------------------------------------------------------------------
// DCNv2 for gfx1250 (MI455X, wave32, WMMA).
//   K0a: transpose x NCHW -> NHWC (x fits in 192MB L2; one-time 16MB)
//   K0b: w_conv fp32 -> bf16, reordered to kdim = kk*256 + c
//   K1 : 27-channel 3x3 offset conv -> om (NCHW x, pixel-coalesced)
//   K2 : bilinear sampling * sigmoid(mask) -> bf16 A [16384][kk*256+c]
//        (contiguous float4 corner reads, contiguous uint4 bf16 writes)
//   K3 : WMMA bf16 GEMM: out[m][o] = A[m][:] . Bt[o][:]  (kdim-order agnostic)
// ---------------------------------------------------------------------------

typedef __attribute__((ext_vector_type(16))) __bf16 v16bf;
typedef __attribute__((ext_vector_type(8)))  float  v8f;

union BF16Frag { v16bf v; uint4 q[2]; };

__device__ __forceinline__ unsigned int f2bf(float f) {
  unsigned int u = __float_as_uint(f);
  u += 0x7FFFu + ((u >> 16) & 1u);      // round-to-nearest-even
  return u >> 16;
}

#define B_ 4
#define C_ 256
#define H_ 64
#define W_ 64
#define O_ 256
#define K_ 9
#define M_TOTAL (B_*H_*W_)   // 16384
#define KDIM    (C_*K_)      // 2304

// ---- K0a: x (B,C,H,W) -> xt (B,H,W,C)
__global__ void x_to_nhwc(const float* __restrict__ x, float* __restrict__ xt) {
  int i = blockIdx.x * blockDim.x + threadIdx.x;   // m*256 + c
  if (i >= B_ * C_ * H_ * W_) return;
  int c   = i & 255;
  int m   = i >> 8;                                // b*4096 + pix
  int b   = m >> 12;
  int pix = m & 4095;
  xt[i] = x[((size_t)(b * C_ + c) << 12) + pix];
}

// ---- K0b: w_conv (O,C,3,3) fp32 -> bf16 Bt[o][kk*256 + c]
__global__ void wconv_reorder_bf16(const float* __restrict__ w,
                                   unsigned short* __restrict__ wbf) {
  int i = blockIdx.x * blockDim.x + threadIdx.x;
  if (i >= O_ * KDIM) return;
  int o  = i / KDIM;
  int r  = i - o * KDIM;
  int kk = r >> 8;
  int c  = r & 255;
  wbf[i] = (unsigned short)f2bf(w[o * KDIM + c * 9 + kk]);
}

// ---- K1: direct 3x3 conv, 27 output channels. One thread per (b,j,h,w).
__global__ void offset_conv(const float* __restrict__ x,
                            const float* __restrict__ w_off,
                            const float* __restrict__ b_off,
                            float* __restrict__ om) {
  int idx = blockIdx.x * blockDim.x + threadIdx.x;   // ((b*27+j)*64+h)*64+w
  if (idx >= B_ * 27 * H_ * W_) return;
  int wp = idx & 63;
  int h  = (idx >> 6) & 63;
  int bj = idx >> 12;
  int j  = bj % 27;
  int b  = bj / 27;

  float acc = b_off[j];
  const float* xb = x + (size_t)b * C_ * H_ * W_;
  const float* wj = w_off + (size_t)j * C_ * 9;
  for (int c = 0; c < C_; ++c) {
    const float* xc = xb + c * H_ * W_;
    const float* wc = wj + c * 9;
#pragma unroll
    for (int kh = 0; kh < 3; ++kh) {
      int y = h - 1 + kh;
      if (y < 0 || y >= H_) continue;
#pragma unroll
      for (int kw = 0; kw < 3; ++kw) {
        int xx = wp - 1 + kw;
        if (xx < 0 || xx >= W_) continue;
        acc = fmaf(xc[y * W_ + xx], wc[kh * 3 + kw], acc);
      }
    }
  }
  om[idx] = acc;
}

// ---- K2: deformable bilinear sampling -> A matrix bf16 [m][kk*256 + c]
__global__ void dcn_sample(const float* __restrict__ xt,   // NHWC
                           const float* __restrict__ om,
                           unsigned short* __restrict__ valA) {
  int t = blockIdx.x * blockDim.x + threadIdx.x;     // m*9 + kk
  if (t >= M_TOTAL * K_) return;
  int kk = t % 9;
  int m  = t / 9;
  int wp = m & 63;
  int h  = (m >> 6) & 63;
  int b  = m >> 12;
  int ky = kk / 3, kx = kk % 3;

  const float* omb = om + (size_t)b * 27 * H_ * W_;
  int pix = h * W_ + wp;
  float offy = omb[(2 * kk)     * (H_ * W_) + pix];
  float offx = omb[(2 * kk + 1) * (H_ * W_) + pix];
  float mr   = omb[(18 + kk)    * (H_ * W_) + pix];
  float mask = 1.0f / (1.0f + __expf(-mr));

  float py = (float)(h - 1 + ky) + offy;
  float px = (float)(wp - 1 + kx) + offx;
  float y0f = floorf(py), x0f = floorf(px);
  float wy = py - y0f, wx = px - x0f;
  int y0 = (int)y0f, x0 = (int)x0f, y1 = y0 + 1, x1 = x0 + 1;

  bool vy0 = (y0 >= 0) && (y0 < H_), vy1 = (y1 >= 0) && (y1 < H_);
  bool vx0 = (x0 >= 0) && (x0 < W_), vx1 = (x1 >= 0) && (x1 < W_);
  int y0c = min(max(y0, 0), H_ - 1), y1c = min(max(y1, 0), H_ - 1);
  int x0c = min(max(x0, 0), W_ - 1), x1c = min(max(x1, 0), W_ - 1);

  float w00 = (1.f - wy) * (1.f - wx) * mask;
  float w01 = (1.f - wy) * wx         * mask;
  float w10 = wy         * (1.f - wx) * mask;
  float w11 = wy         * wx         * mask;
  if (!(vy0 && vx0)) w00 = 0.f;
  if (!(vy0 && vx1)) w01 = 0.f;
  if (!(vy1 && vx0)) w10 = 0.f;
  if (!(vy1 && vx1)) w11 = 0.f;

  const size_t base = (size_t)b * (H_ * W_ * C_);
  const float4* c00 = (const float4*)(xt + base + (y0c * W_ + x0c) * C_);
  const float4* c01 = (const float4*)(xt + base + (y0c * W_ + x1c) * C_);
  const float4* c10 = (const float4*)(xt + base + (y1c * W_ + x0c) * C_);
  const float4* c11 = (const float4*)(xt + base + (y1c * W_ + x1c) * C_);

  uint4* dst = (uint4*)(valA + (size_t)m * KDIM + kk * C_);
#pragma unroll 4
  for (int c8 = 0; c8 < C_ / 8; ++c8) {
    float4 a0 = c00[c8 * 2],     b0 = c01[c8 * 2];
    float4 g0 = c10[c8 * 2],     d0 = c11[c8 * 2];
    float4 a1 = c00[c8 * 2 + 1], b1 = c01[c8 * 2 + 1];
    float4 g1 = c10[c8 * 2 + 1], d1 = c11[c8 * 2 + 1];
    float v0 = w00 * a0.x + w01 * b0.x + w10 * g0.x + w11 * d0.x;
    float v1 = w00 * a0.y + w01 * b0.y + w10 * g0.y + w11 * d0.y;
    float v2 = w00 * a0.z + w01 * b0.z + w10 * g0.z + w11 * d0.z;
    float v3 = w00 * a0.w + w01 * b0.w + w10 * g0.w + w11 * d0.w;
    float v4 = w00 * a1.x + w01 * b1.x + w10 * g1.x + w11 * d1.x;
    float v5 = w00 * a1.y + w01 * b1.y + w10 * g1.y + w11 * d1.y;
    float v6 = w00 * a1.z + w01 * b1.z + w10 * g1.z + w11 * d1.z;
    float v7 = w00 * a1.w + w01 * b1.w + w10 * g1.w + w11 * d1.w;
    uint4 q;
    q.x = f2bf(v0) | (f2bf(v1) << 16);
    q.y = f2bf(v2) | (f2bf(v3) << 16);
    q.z = f2bf(v4) | (f2bf(v5) << 16);
    q.w = f2bf(v6) | (f2bf(v7) << 16);
    dst[c8] = q;
  }
}

// ---- K3: bf16 WMMA GEMM. Block tile 128M x 64N, 8 waves (4x2), wave 32x32.
#define MT 128
#define NT 64
#define KT 32

__global__ __launch_bounds__(256) void dcn_gemm(
    const unsigned short* __restrict__ A,    // [16384][2304] bf16
    const unsigned short* __restrict__ Bt,   // [256][2304]   bf16
    float* __restrict__ out) {               // (B,O,H,W)
  __shared__ __align__(16) unsigned short lA[MT][KT];
  __shared__ __align__(16) unsigned short lB[NT][KT];

  const int tid  = threadIdx.x;
  const int lane = tid & 31;
  const int wave = tid >> 5;
  const int wm = wave >> 1;                  // 0..3
  const int wn = wave & 1;                   // 0..1
  const int mBase = blockIdx.x * MT;
  const int nBase = blockIdx.y * NT;

  v8f acc[2][2] = {};

  // LDS fill assignments (A: 512 uint4 -> 2/thread; B: 256 uint4 -> 1/thread)
  const int aRow   = tid >> 1;
  const int aChunk = (tid & 1) * 2;          // uint4 chunk within 32-elem row
  const int bRow   = tid >> 2;
  const int bChunk = tid & 3;

  const int mrow  = lane & 15;
  const int kOffA = (lane < 16) ? 0 : 8;     // A frag: K {0..7,16..23} / {8..15,24..31}
  const int kOffB = (lane < 16) ? 0 : 8;     // B frag: K {0..7,16..23} / {8..15,24..31}

  for (int k0 = 0; k0 < KDIM; k0 += KT) {
    const uint4* ga =
        reinterpret_cast<const uint4*>(A + (size_t)(mBase + aRow) * KDIM + k0) + aChunk;
    uint4 a0 = ga[0];
    uint4 a1 = ga[1];
    uint4 bv =
        reinterpret_cast<const uint4*>(Bt + (size_t)(nBase + bRow) * KDIM + k0)[bChunk];

    __syncthreads();   // previous iteration's fragments consumed
    *reinterpret_cast<uint4*>(&lA[aRow][aChunk * 8])     = a0;
    *reinterpret_cast<uint4*>(&lA[aRow][aChunk * 8 + 8]) = a1;
    *reinterpret_cast<uint4*>(&lB[bRow][bChunk * 8])     = bv;
    __syncthreads();

    BF16Frag af[2], bf[2];
#pragma unroll
    for (int ti = 0; ti < 2; ++ti) {
      const unsigned short* p = &lA[wm * 32 + ti * 16 + mrow][0];
      af[ti].q[0] = *reinterpret_cast<const uint4*>(p + kOffA);
      af[ti].q[1] = *reinterpret_cast<const uint4*>(p + kOffA + 16);
    }
#pragma unroll
    for (int tj = 0; tj < 2; ++tj) {
      const unsigned short* p = &lB[wn * 32 + tj * 16 + mrow][0];
      bf[tj].q[0] = *reinterpret_cast<const uint4*>(p + kOffB);
      bf[tj].q[1] = *reinterpret_cast<const uint4*>(p + kOffB + 16);
    }
#pragma unroll
    for (int ti = 0; ti < 2; ++ti)
#pragma unroll
      for (int tj = 0; tj < 2; ++tj)
        acc[ti][tj] = __builtin_amdgcn_wmma_f32_16x16x32_bf16(
            false, af[ti].v, false, bf[tj].v, (short)0, acc[ti][tj], false, false);
  }

  // Epilogue: C/D layout -> out[b][o][h][w]
#pragma unroll
  for (int ti = 0; ti < 2; ++ti) {
    int mTile = mBase + wm * 32 + ti * 16;
#pragma unroll
    for (int tj = 0; tj < 2; ++tj) {
      int col = nBase + wn * 32 + tj * 16 + (lane & 15);
#pragma unroll
      for (int r = 0; r < 8; ++r) {
        int row = mTile + ((lane >> 4) << 3) + r;   // M = r (lanes 0-15) / r+8
        int b  = row >> 12;
        int hw = row & 4095;
        out[((size_t)(b * O_ + col) << 12) + hw] = acc[ti][tj][r];
      }
    }
  }
}

// ---------------------------------------------------------------------------
extern "C" void kernel_launch(void* const* d_in, const int* in_sizes, int n_in,
                              void* d_out, int out_size, void* d_ws, size_t ws_size,
                              hipStream_t stream) {
  const float* x      = (const float*)d_in[0];
  const float* w_off  = (const float*)d_in[1];
  const float* b_off  = (const float*)d_in[2];
  const float* w_conv = (const float*)d_in[3];
  float* out = (float*)d_out;

  // Workspace layout (each region 256B aligned):
  //   valA : 16384*2304 bf16 = 75,497,472 B
  //   xt   : 4*64*64*256 f32 = 16,777,216 B
  //   om   : 4*27*4096  f32  =  1,769,472 B
  //   wbf  : 256*2304   bf16 =  1,179,648 B
  char* ws = (char*)d_ws;
  size_t valBytes = (size_t)M_TOTAL * KDIM * sizeof(unsigned short);
  size_t xtBytes  = (size_t)B_ * C_ * H_ * W_ * sizeof(float);
  size_t omBytes  = (size_t)B_ * 27 * H_ * W_ * sizeof(float);
  unsigned short* valA = (unsigned short*)ws;
  float* xt = (float*)(ws + ((valBytes + 255) & ~(size_t)255));
  float* om = (float*)((char*)xt + ((xtBytes + 255) & ~(size_t)255));
  unsigned short* wbf =
      (unsigned short*)((char*)om + ((omBytes + 255) & ~(size_t)255));

  x_to_nhwc<<<(B_ * C_ * H_ * W_ + 255) / 256, 256, 0, stream>>>(x, xt);
  wconv_reorder_bf16<<<(O_ * KDIM + 255) / 256, 256, 0, stream>>>(w_conv, wbf);
  offset_conv<<<(B_ * 27 * H_ * W_ + 255) / 256, 256, 0, stream>>>(x, w_off, b_off, om);
  dcn_sample<<<(M_TOTAL * K_ + 255) / 256, 256, 0, stream>>>(xt, om, valA);

  dim3 g(M_TOTAL / MT, O_ / NT);   // 128 x 4 workgroups
  dcn_gemm<<<g, 256, 0, stream>>>(valA, wbf, out);
}
